// SyntaxMultiAttentionGAT_87144886436423
// MI455X (gfx1250) — matrix operand order repaired and checked by reading, not compiled, not run on hardware
//
#include <hip/hip_runtime.h>
#include <hip/hip_bf16.h>

// ---------------------------------------------------------------------------
// Problem constants (SyntaxMultiAttentionGAT reference)
// ---------------------------------------------------------------------------
constexpr int Bb  = 4;
constexpr int Ll  = 96;
constexpr int Dd  = 256;
constexpr int Hh  = 8;
constexpr int HDd = 32;                 // head dim
constexpr int FFf = 1024;
constexpr int Nn  = Bb * Ll * Ll;       // 36864 rows

// ---------------------------------------------------------------------------
// WMMA types (CDNA5 gfx1250, wave32)
// ---------------------------------------------------------------------------
typedef __attribute__((ext_vector_type(16))) __bf16       v16bf;
typedef __attribute__((ext_vector_type(8)))  float        v8f;
typedef __attribute__((ext_vector_type(4)))  unsigned int v4u;

union Frag16 {
    v16bf          v;
    v4u            u[2];
    unsigned short s[16];
};

__device__ __forceinline__ v8f wmma_bf16(v16bf a, v16bf b, v8f c) {
    // 8 args: (neg_a, A, neg_b, B, c_mod, C, reuse_a, reuse_b)
    return __builtin_amdgcn_wmma_f32_16x16x32_bf16(
        false, a, false, b, (short)0, c, false, false);
}

__device__ __forceinline__ unsigned short f2bf(float f) {
    unsigned int u = __builtin_bit_cast(unsigned int, f);
    u += 0x7FFFu + ((u >> 16) & 1u);        // round-to-nearest-even
    return (unsigned short)(u >> 16);
}

// ---------------------------------------------------------------------------
// Weight convert + transpose:  dst[n*K + k] = bf16(src[k*N + n])
// ---------------------------------------------------------------------------
__global__ void cvt_w_kernel(const float* __restrict__ src,
                             unsigned short* __restrict__ dst,
                             int K, int Nc) {
    int idx = blockIdx.x * blockDim.x + threadIdx.x;
    if (idx >= K * Nc) return;
    int n = idx / K;
    int k = idx - n * K;
    dst[idx] = f2bf(src[(size_t)k * Nc + n]);
}

// fp32 -> bf16 elementwise
__global__ void cvt_f_kernel(const float* __restrict__ src,
                             unsigned short* __restrict__ dst, int count) {
    int idx = blockIdx.x * blockDim.x + threadIdx.x;
    if (idx < count) dst[idx] = f2bf(src[idx]);
}

// ---------------------------------------------------------------------------
// Generic bf16 GEMM:  out[M,Nc] = A[M,K] @ W  (+ bias, epilogue)
//   A : row-major bf16, leading dim lda
//   Wt: bf16 weights stored TRANSPOSED: Wt[n*K + k]
// Block = 256 threads = 8 waves; each wave owns one 16x16 output tile.
// EPI: 0 = bf16 out (bias)         1 = bf16 out (bias + relu)
//      2 = f32 out  (bias + add)   3 = f32 out  (bias + add)
// ---------------------------------------------------------------------------
template <int EPI>
__global__ __launch_bounds__(256) void gemm_bf16_kernel(
    const unsigned short* __restrict__ A, int lda,
    const unsigned short* __restrict__ Wt,
    const float* __restrict__ bias,
    const float* __restrict__ addsrc,          // residual [M,Nc] (EPI>=2)
    unsigned short* __restrict__ outbf,
    float* __restrict__ outf,
    int M, int Nc, int K) {
    const int lane  = threadIdx.x & 31;
    const int wv    = threadIdx.x >> 5;
    const int mbase = (blockIdx.y * 8 + wv) * 16;
    const int nbase = blockIdx.x * 16;
    if (mbase >= M) return;                    // wave-uniform exit

    const int ml = lane & 15;
    const int hi = lane >> 4;

    // A fragment: row m = mbase+ml; K elems {hi*8+0..7, 16+hi*8+0..7}
    const unsigned short* ap = A + (size_t)(mbase + ml) * lda + hi * 8;
    // B fragment: col n = nbase+ml; K elems {hi*8+0..7, 16+hi*8+0..7}
    const unsigned short* bp = Wt + (size_t)(nbase + ml) * K + hi * 8;

    v8f acc;
#pragma unroll
    for (int i = 0; i < 8; ++i) acc[i] = 0.0f;

    for (int k0 = 0; k0 < K; k0 += 32) {
        Frag16 fa, fb;
        fa.u[0] = *(const v4u*)(ap);
        fa.u[1] = *(const v4u*)(ap + 16);
        fb.u[0] = *(const v4u*)(bp);
        fb.u[1] = *(const v4u*)(bp + 16);
        acc = wmma_bf16(fa.v, fb.v, acc);
        ap += 32;
        bp += 32;
    }

    const float bcol = bias[nbase + ml];
#pragma unroll
    for (int r = 0; r < 8; ++r) {
        const int row = mbase + hi * 8 + r;    // C/D layout: M = r + hi*8
        const int col = nbase + ml;
        float v = acc[r] + bcol;
        if constexpr (EPI == 1) v = v > 0.0f ? v : 0.0f;
        if constexpr (EPI >= 2) v += addsrc[(size_t)row * Nc + col];
        if constexpr (EPI <= 1)
            outbf[(size_t)row * Nc + col] = f2bf(v);
        else
            outf[(size_t)row * Nc + col] = v;
    }
}

// ---------------------------------------------------------------------------
// Fused attention per (b, h, x):  scores -> +dep bias -> softmax -> ctx
// Q/K/V are [N,256] bf16; the 96-row slab for (b,x) is contiguous.
// Block = 192 threads = 6 waves; wave w owns q-rows [16w, 16w+16).
// ---------------------------------------------------------------------------
__global__ __launch_bounds__(192) void attn_kernel(
    const unsigned short* __restrict__ Q,
    const unsigned short* __restrict__ Km,
    const unsigned short* __restrict__ V,
    const float* __restrict__ dep,             // [B,L,L]
    const float* __restrict__ Wdep,            // [H]
    const float* __restrict__ bdep,            // [H]
    unsigned short* __restrict__ Ctx) {        // [N,256] bf16
    __shared__ float          sS[Ll * Ll];     // fp32 scores   (36 KB)
    __shared__ unsigned short sP[Ll * Ll];     // bf16 probs    (18 KB)

    const int x  = blockIdx.x;
    const int h  = blockIdx.y;
    const int b  = blockIdx.z;
    const int lane = threadIdx.x & 31;
    const int wv   = threadIdx.x >> 5;
    const int ml = lane & 15;
    const int hi = lane >> 4;
    const int rowbase = (b * Ll + x) * Ll;     // row into [N,256]
    const int mbase   = wv * 16;
    const float scale = 0.17677669529663687f;  // 1/sqrt(32)
    const float wd = Wdep[h];
    const float bd = bdep[h];

    // ---- scores = (q @ k^T) * scale + dep bias -----------------------------
    Frag16 fa;
    {
        const unsigned short* qp =
            Q + (size_t)(rowbase + mbase + ml) * Dd + h * HDd + hi * 8;
        fa.u[0] = *(const v4u*)(qp);
        fa.u[1] = *(const v4u*)(qp + 16);
    }
#pragma unroll
    for (int ct = 0; ct < 6; ++ct) {
        Frag16 fb;
        const unsigned short* kp =
            Km + (size_t)(rowbase + ct * 16 + ml) * Dd + h * HDd + hi * 8;
        fb.u[0] = *(const v4u*)(kp);
        fb.u[1] = *(const v4u*)(kp + 16);
        v8f acc;
#pragma unroll
        for (int i = 0; i < 8; ++i) acc[i] = 0.0f;
        acc = wmma_bf16(fa.v, fb.v, acc);      // K = 32 = hd: one WMMA/tile

        const int   col = ct * 16 + ml;
        const float bias = dep[(size_t)rowbase + col] * wd + bd;
#pragma unroll
        for (int r = 0; r < 8; ++r)
            sS[(mbase + hi * 8 + r) * Ll + col] = acc[r] * scale + bias;
    }
    __syncthreads();

    // ---- row softmax (threads 0..95 own one row each) ----------------------
    if (threadIdx.x < Ll) {
        float* rowp = &sS[threadIdx.x * Ll];
        float mx = rowp[0];
        for (int i = 1; i < Ll; ++i) mx = fmaxf(mx, rowp[i]);
        float sum = 0.0f;
        for (int i = 0; i < Ll; ++i) {
            float e = __expf(rowp[i] - mx);
            rowp[i] = e;
            sum += e;
        }
        const float inv = 1.0f / sum;
        for (int i = 0; i < Ll; ++i)
            sP[threadIdx.x * Ll + i] = f2bf(rowp[i] * inv);
    }
    __syncthreads();

    // ---- ctx = probs @ v  (wave: 16 rows x 32 cols = 2 tiles, K=96) --------
#pragma unroll
    for (int ct = 0; ct < 2; ++ct) {
        v8f acc;
#pragma unroll
        for (int i = 0; i < 8; ++i) acc[i] = 0.0f;
#pragma unroll
        for (int k0 = 0; k0 < Ll; k0 += 32) {
            Frag16 pa;
            const unsigned short* pp = &sP[(mbase + ml) * Ll + k0 + hi * 8];
            pa.u[0] = *(const v4u*)(pp);
            pa.u[1] = *(const v4u*)(pp + 16);
            Frag16 vb;
            const int n  = ct * 16 + ml;
            const int kb = k0 + hi * 8;
#pragma unroll
            for (int i = 0; i < 16; ++i)
                vb.s[i] = V[(size_t)(rowbase + kb + (i < 8 ? i : i + 8)) * Dd + h * HDd + n];
            acc = wmma_bf16(pa.v, vb.v, acc);
        }
#pragma unroll
        for (int r = 0; r < 8; ++r) {
            const int row = rowbase + mbase + hi * 8 + r;
            const int col = h * HDd + ct * 16 + ml;
            Ctx[(size_t)row * Dd + col] = f2bf(acc[r]);
        }
    }
}

// ---------------------------------------------------------------------------
// LayerNorm over D=256, one row per block (256 threads)
// ---------------------------------------------------------------------------
__global__ __launch_bounds__(256) void ln_kernel(
    const float* __restrict__ x,
    const float* __restrict__ g,
    const float* __restrict__ bta,
    float* __restrict__ outf,
    unsigned short* __restrict__ outbf) {      // may be nullptr
    __shared__ float red[256];
    const int row = blockIdx.x;
    const int t   = threadIdx.x;
    const float v = x[(size_t)row * Dd + t];

    red[t] = v;
    __syncthreads();
    for (int s = 128; s > 0; s >>= 1) {
        if (t < s) red[t] += red[t + s];
        __syncthreads();
    }
    const float mean = red[0] * (1.0f / 256.0f);
    __syncthreads();

    const float d = v - mean;
    red[t] = d * d;
    __syncthreads();
    for (int s = 128; s > 0; s >>= 1) {
        if (t < s) red[t] += red[t + s];
        __syncthreads();
    }
    const float var = red[0] * (1.0f / 256.0f);

    const float y = d * rsqrtf(var + 1e-5f) * g[t] + bta[t];
    outf[(size_t)row * Dd + t] = y;
    if (outbf) outbf[(size_t)row * Dd + t] = f2bf(y);
}

// ---------------------------------------------------------------------------
// Workspace layout (bytes). Aliases exploit liveness:
//   TMP1(f32, pre-LN1)  aliases [QBF..KBF]   (Q/K dead after attention)
//   OUTF(f32, LN1 out)  aliases [VBF..CTX]   (V dead; Ctx dead after out-proj)
//   OUTBF(bf16)         aliases XBF          (X dead after QKV)
//   TMP2(f32, pre-LN2)  aliases [QBF..KBF]   (TMP1 dead after LN1)
// ---------------------------------------------------------------------------
constexpr size_t SZ_W256 = (size_t)Dd * Dd * 2;     // 128 KB
constexpr size_t SZ_WFF  = (size_t)Dd * FFf * 2;    // 512 KB
constexpr size_t SZ_NBF  = (size_t)Nn * Dd * 2;     // 18.9 MB

constexpr size_t OFF_WQT = 0;
constexpr size_t OFF_WKT = OFF_WQT + SZ_W256;
constexpr size_t OFF_WVT = OFF_WKT + SZ_W256;
constexpr size_t OFF_WOT = OFF_WVT + SZ_W256;
constexpr size_t OFF_W1T = OFF_WOT + SZ_W256;
constexpr size_t OFF_W2T = OFF_W1T + SZ_WFF;
constexpr size_t OFF_XBF = OFF_W2T + SZ_WFF;
constexpr size_t OFF_QBF = OFF_XBF + SZ_NBF;
constexpr size_t OFF_KBF = OFF_QBF + SZ_NBF;
constexpr size_t OFF_VBF = OFF_KBF + SZ_NBF;
constexpr size_t OFF_CTX = OFF_VBF + SZ_NBF;
constexpr size_t OFF_FF1 = OFF_CTX + SZ_NBF;        // chunk buffer (18.9 MB)

constexpr int NCH = 4;
constexpr int MCH = Nn / NCH;                       // 9216 rows per FFN chunk

extern "C" void kernel_launch(void* const* d_in, const int* in_sizes, int n_in,
                              void* d_out, int out_size, void* d_ws, size_t ws_size,
                              hipStream_t stream) {
    (void)in_sizes; (void)n_in; (void)out_size; (void)ws_size;

    const float* table = (const float*)d_in[0];
    const float* dep   = (const float*)d_in[1];
    const float* Wq    = (const float*)d_in[2];
    const float* bq    = (const float*)d_in[3];
    const float* Wk    = (const float*)d_in[4];
    const float* bk    = (const float*)d_in[5];
    const float* Wv    = (const float*)d_in[6];
    const float* bv    = (const float*)d_in[7];
    const float* Wo    = (const float*)d_in[8];
    const float* bo    = (const float*)d_in[9];
    const float* Wdep  = (const float*)d_in[10];
    const float* bdep  = (const float*)d_in[11];
    const float* ln1g  = (const float*)d_in[12];
    const float* ln1b  = (const float*)d_in[13];
    const float* W1    = (const float*)d_in[14];
    const float* b1    = (const float*)d_in[15];
    const float* W2    = (const float*)d_in[16];
    const float* b2    = (const float*)d_in[17];
    const float* ln2g  = (const float*)d_in[18];
    const float* ln2b  = (const float*)d_in[19];

    char* ws = (char*)d_ws;
    unsigned short* WQT = (unsigned short*)(ws + OFF_WQT);
    unsigned short* WKT = (unsigned short*)(ws + OFF_WKT);
    unsigned short* WVT = (unsigned short*)(ws + OFF_WVT);
    unsigned short* WOT = (unsigned short*)(ws + OFF_WOT);
    unsigned short* W1T = (unsigned short*)(ws + OFF_W1T);
    unsigned short* W2T = (unsigned short*)(ws + OFF_W2T);
    unsigned short* XBF = (unsigned short*)(ws + OFF_XBF);
    unsigned short* QBF = (unsigned short*)(ws + OFF_QBF);
    unsigned short* KBF = (unsigned short*)(ws + OFF_KBF);
    unsigned short* VBF = (unsigned short*)(ws + OFF_VBF);
    unsigned short* CTX = (unsigned short*)(ws + OFF_CTX);
    unsigned short* FF1 = (unsigned short*)(ws + OFF_FF1);
    float*          TMP1  = (float*)(ws + OFF_QBF);   // alias
    float*          OUTF  = (float*)(ws + OFF_VBF);   // alias
    unsigned short* OUTBF = XBF;                      // alias
    float*          TMP2  = (float*)(ws + OFF_QBF);   // alias

    // 1) weights -> bf16, transposed to [n][k]
    cvt_w_kernel<<<(Dd * Dd + 255) / 256, 256, 0, stream>>>(Wq, WQT, Dd, Dd);
    cvt_w_kernel<<<(Dd * Dd + 255) / 256, 256, 0, stream>>>(Wk, WKT, Dd, Dd);
    cvt_w_kernel<<<(Dd * Dd + 255) / 256, 256, 0, stream>>>(Wv, WVT, Dd, Dd);
    cvt_w_kernel<<<(Dd * Dd + 255) / 256, 256, 0, stream>>>(Wo, WOT, Dd, Dd);
    cvt_w_kernel<<<(Dd * FFf + 255) / 256, 256, 0, stream>>>(W1, W1T, Dd, FFf);
    cvt_w_kernel<<<(Dd * FFf + 255) / 256, 256, 0, stream>>>(W2, W2T, FFf, Dd);

    // 2) table -> bf16 activations
    cvt_f_kernel<<<(Nn * Dd + 255) / 256, 256, 0, stream>>>(table, XBF, Nn * Dd);

    // 3) Q/K/V projections (bf16 WMMA, fp32 accum)
    dim3 gP(Dd / 16, Nn / 128);
    gemm_bf16_kernel<0><<<gP, 256, 0, stream>>>(XBF, Dd, WQT, bq, nullptr, QBF, nullptr, Nn, Dd, Dd);
    gemm_bf16_kernel<0><<<gP, 256, 0, stream>>>(XBF, Dd, WKT, bk, nullptr, KBF, nullptr, Nn, Dd, Dd);
    gemm_bf16_kernel<0><<<gP, 256, 0, stream>>>(XBF, Dd, WVT, bv, nullptr, VBF, nullptr, Nn, Dd, Dd);

    // 4) attention per (b, h, x)
    attn_kernel<<<dim3(Ll, Hh, Bb), 192, 0, stream>>>(QBF, KBF, VBF, dep, Wdep, bdep, CTX);

    // 5) out-proj + table residual (fp32 out), then LN1
    gemm_bf16_kernel<2><<<gP, 256, 0, stream>>>(CTX, Dd, WOT, bo, table, nullptr, TMP1, Nn, Dd, Dd);
    ln_kernel<<<Nn, 256, 0, stream>>>(TMP1, ln1g, ln1b, OUTF, OUTBF);

    // 6) FFN in row chunks (bounds the 1024-wide hidden buffer)
    for (int c = 0; c < NCH; ++c) {
        const size_t ro = (size_t)c * MCH;
        gemm_bf16_kernel<1><<<dim3(FFf / 16, MCH / 128), 256, 0, stream>>>(
            OUTBF + ro * Dd, Dd, W1T, b1, nullptr, FF1, nullptr, MCH, FFf, Dd);
        gemm_bf16_kernel<3><<<dim3(Dd / 16, MCH / 128), 256, 0, stream>>>(
            FF1, FFf, W2T, b2, OUTF + ro * Dd, nullptr, TMP2 + ro * Dd, MCH, Dd, FFf);
    }

    // 7) LN2 -> final output (fp32)
    ln_kernel<<<Nn, 256, 0, stream>>>(TMP2, ln2g, ln2b, (float*)d_out, nullptr);
}
